// CausalSelfAttention_61005715472498
// MI455X (gfx1250) — matrix operand order repaired and checked by reading, not compiled, not run on hardware
//
#include <hip/hip_runtime.h>

typedef __attribute__((ext_vector_type(16))) __bf16 v16bf;
typedef __attribute__((ext_vector_type(8)))  __bf16 v8bf;
typedef __attribute__((ext_vector_type(4)))  __bf16 v4bf;
typedef __attribute__((ext_vector_type(8)))  float  v8f;

static __device__ __forceinline__ v8f wmma_bf16(v16bf a, v16bf b, v8f c) {
    return __builtin_amdgcn_wmma_f32_16x16x32_bf16(false, a, false, b, (short)0, c, false, false);
}
static __device__ __forceinline__ v16bf cat8(v8bf a, v8bf b) {
    return __builtin_shufflevector(a, b, 0,1,2,3,4,5,6,7,8,9,10,11,12,13,14,15);
}

// Problem constants
#define BB 2
#define TT 2048
#define CC 1024
#define HH 16
#define HS 64
#define MM (BB*TT)      // 4096
#define N1 (3*CC)       // 3072
#define LSTR 80         // LDS row stride (bf16 elems): 160B rows, 32B-aligned
#define KBLK 64

// ---------------------------------------------------------------------------
// Kernel 0: one-time weight conversion: Wat[n][k] = bf16(w_attn[k][n]),
//           Wpt[n][k] = bf16(w_proj[k][n])
// ---------------------------------------------------------------------------
__global__ __launch_bounds__(256) void k_convw(const float* __restrict__ wa,
                                               const float* __restrict__ wp,
                                               __bf16* __restrict__ Wat,
                                               __bf16* __restrict__ Wpt) {
    int idx = blockIdx.x * 256 + threadIdx.x;       // 0 .. 4M-1
    const int NA = N1 * CC;                          // 3M
    if (idx < NA) {
        int n = idx >> 10, k = idx & 1023;
        Wat[idx] = (__bf16)wa[(size_t)k * N1 + n];
    } else {
        int o = idx - NA;
        int n = o >> 10, k = o & 1023;
        Wpt[o] = (__bf16)wp[(size_t)k * CC + n];
    }
}

// ---------------------------------------------------------------------------
// Kernel 1: qkv = x @ w_attn + b_attn ; scatter to Q[B,H,T,64], K[B,H,T,64],
//           Vt[B,H,64,T] (all bf16). A: fp32 on the fly; B: pre-transposed bf16.
// ---------------------------------------------------------------------------
__global__ __launch_bounds__(256) void k_qkv(const float* __restrict__ x,
                                             const __bf16* __restrict__ Wat,
                                             const float* __restrict__ bias,
                                             __bf16* __restrict__ Qb,
                                             __bf16* __restrict__ Kb,
                                             __bf16* __restrict__ Vt) {
    __shared__ __attribute__((aligned(32))) __bf16 As[128 * LSTR];
    __shared__ __attribute__((aligned(32))) __bf16 Bs[128 * LSTR];
    const int tid  = threadIdx.x;
    const int lane = tid & 31;
    const int wave = tid >> 5;
    const int l16  = lane & 15;
    const int hi   = lane >> 4;
    const int bm = blockIdx.y * 128;
    const int bn = blockIdx.x * 128;
    const int wm = (wave & 1) * 64;
    const int wn = (wave >> 1) * 32;

    const float bias0 = bias[bn + wn + l16];
    const float bias1 = bias[bn + wn + 16 + l16];

    v8f acc[4][2];
#pragma unroll
    for (int i = 0; i < 4; ++i)
#pragma unroll
        for (int j = 0; j < 2; ++j)
#pragma unroll
            for (int e = 0; e < 8; ++e) acc[i][j][e] = 0.0f;

    struct Stage { float4 a[8]; v16bf b[2]; };
    Stage cur, nxt;

    auto ld = [&](Stage& s, int kb) {
#pragma unroll
        for (int j = 0; j < 8; ++j) {              // A: 128x64 fp32 = 2048 float4
            int idx = tid + j * 256;
            int row = idx >> 4, c4 = (idx & 15) << 2;
            s.a[j] = *(const float4*)(x + (size_t)(bm + row) * CC + kb + c4);
        }
#pragma unroll
        for (int j = 0; j < 2; ++j) {              // B: 128x64 bf16 = 512 x 32B
            int idx = tid + j * 256;
            int row = idx >> 2, c16 = (idx & 3) << 4;
            s.b[j] = *(const v16bf*)(Wat + (size_t)(bn + row) * CC + kb + c16);
        }
    };
    auto st = [&](Stage& s) {
#pragma unroll
        for (int j = 0; j < 8; ++j) {
            int idx = tid + j * 256;
            int row = idx >> 4, c4 = (idx & 15) << 2;
            v4bf p;
            p[0] = (__bf16)s.a[j].x; p[1] = (__bf16)s.a[j].y;
            p[2] = (__bf16)s.a[j].z; p[3] = (__bf16)s.a[j].w;
            *(v4bf*)&As[row * LSTR + c4] = p;      // ds_store_b64
        }
#pragma unroll
        for (int j = 0; j < 2; ++j) {
            int idx = tid + j * 256;
            int row = idx >> 2, c16 = (idx & 3) << 4;
            *(v16bf*)&Bs[row * LSTR + c16] = s.b[j];
        }
    };

    ld(cur, 0);
    for (int it = 0; it < CC / KBLK; ++it) {
        st(cur);
        __syncthreads();
        if (it < CC / KBLK - 1) ld(nxt, (it + 1) * KBLK);   // overlap with WMMA
#pragma unroll
        for (int ks = 0; ks < 2; ++ks) {
            v16bf af[4], bfr[2];
#pragma unroll
            for (int mt = 0; mt < 4; ++mt) {
                const __bf16* p = &As[(wm + mt * 16 + l16) * LSTR + ks * 32 + hi * 16];
                af[mt] = cat8(*(const v8bf*)p, *(const v8bf*)(p + 8));
            }
#pragma unroll
            for (int nt = 0; nt < 2; ++nt)
                bfr[nt] = *(const v16bf*)&Bs[(wn + nt * 16 + l16) * LSTR + ks * 32 + hi * 16];
#pragma unroll
            for (int mt = 0; mt < 4; ++mt)
#pragma unroll
                for (int nt = 0; nt < 2; ++nt)
                    acc[mt][nt] = wmma_bf16(af[mt], bfr[nt], acc[mt][nt]);
        }
        __syncthreads();
        cur = nxt;
    }

    // Epilogue: bias + scatter to Q / K / Vt (bf16)
#pragma unroll
    for (int mt = 0; mt < 4; ++mt)
#pragma unroll
        for (int nt = 0; nt < 2; ++nt)
#pragma unroll
            for (int e = 0; e < 8; ++e) {
                int grow = bm + wm + mt * 16 + e + hi * 8;   // 0..4095
                int gcol = bn + wn + nt * 16 + l16;          // 0..3071
                float val = acc[mt][nt][e] + (nt ? bias1 : bias0);
                int sec = gcol >> 10;
                int c = gcol & 1023;
                int h = c >> 6, d = c & 63;
                int b = grow >> 11, t = grow & 2047;
                size_t bh = (size_t)(b * HH + h);
                if (sec == 0)      Qb[(bh * TT + t) * HS + d] = (__bf16)val;
                else if (sec == 1) Kb[(bh * TT + t) * HS + d] = (__bf16)val;
                else               Vt[(bh * HS + d) * TT + t] = (__bf16)val;
            }
}

// ---------------------------------------------------------------------------
// Kernel 2: flash attention. One wave handles one 16-row Q block of one (b,h).
// ---------------------------------------------------------------------------
__global__ __launch_bounds__(128) void k_attn(const __bf16* __restrict__ Qb,
                                              const __bf16* __restrict__ Kb,
                                              const __bf16* __restrict__ Vt,
                                              __bf16* __restrict__ Y) {
    __shared__ __attribute__((aligned(32))) __bf16 Plds[4][16 * 40];
    const int tid  = threadIdx.x;
    const int lane = tid & 31;
    const int wave = tid >> 5;
    const int l16  = lane & 15;
    const int hi   = lane >> 4;
    const int bh    = blockIdx.y;
    const int qblk  = blockIdx.x * 4 + wave;
    const int qbase = qblk * 16;

    const __bf16* Qp = Qb + (size_t)bh * TT * HS;
    const __bf16* Kp = Kb + (size_t)bh * TT * HS;
    const __bf16* Vp = Vt + (size_t)bh * HS * TT;

    v16bf qf[2];
#pragma unroll
    for (int ks = 0; ks < 2; ++ks) {
        const __bf16* p = Qp + (size_t)(qbase + l16) * HS + ks * 32 + hi * 16;
        qf[ks] = cat8(*(const v8bf*)p, *(const v8bf*)(p + 8));
    }

    float m[8], l[8];
    v8f o[4];
#pragma unroll
    for (int e = 0; e < 8; ++e) { m[e] = -3.0e38f; l[e] = 0.0f; }
#pragma unroll
    for (int nt = 0; nt < 4; ++nt)
#pragma unroll
        for (int e = 0; e < 8; ++e) o[nt][e] = 0.0f;

    const int jend = qbase + 16;
    for (int jb = 0; jb < jend; jb += 32) {
        v8f s[2];
#pragma unroll
        for (int nt = 0; nt < 2; ++nt) {
#pragma unroll
            for (int e = 0; e < 8; ++e) s[nt][e] = 0.0f;
#pragma unroll
            for (int ks = 0; ks < 2; ++ks) {
                const __bf16* p = Kp + (size_t)(jb + nt * 16 + l16) * HS + ks * 32 + hi * 16;
                v16bf kf = *(const v16bf*)p;
                s[nt] = wmma_bf16(qf[ks], kf, s[nt]);
            }
        }
        float rmax[8];
#pragma unroll
        for (int e = 0; e < 8; ++e) {
            int grow = qbase + e + hi * 8;
            float rm = -3.0e38f;
#pragma unroll
            for (int nt = 0; nt < 2; ++nt) {
                int gcol = jb + nt * 16 + l16;
                float v = s[nt][e] * 0.125f;             // 1/sqrt(64)
                v = (gcol <= grow) ? v : -3.0e38f;
                s[nt][e] = v;
                rm = fmaxf(rm, v);
            }
#pragma unroll
            for (int off = 1; off < 16; off <<= 1)
                rm = fmaxf(rm, __shfl_xor(rm, off, 32));
            rmax[e] = rm;
        }
#pragma unroll
        for (int e = 0; e < 8; ++e) {
            float mn = fmaxf(m[e], rmax[e]);
            float alpha = __expf(m[e] - mn);
            m[e] = mn;
            l[e] *= alpha;
#pragma unroll
            for (int nt = 0; nt < 2; ++nt) s[nt][e] = __expf(s[nt][e] - mn);
#pragma unroll
            for (int nt = 0; nt < 4; ++nt) o[nt][e] *= alpha;
            float rs = s[0][e] + s[1][e];
#pragma unroll
            for (int off = 1; off < 16; off <<= 1)
                rs += __shfl_xor(rs, off, 32);
            l[e] += rs;
        }
        __bf16* pl = &Plds[wave][0];
#pragma unroll
        for (int nt = 0; nt < 2; ++nt)
#pragma unroll
            for (int e = 0; e < 8; ++e)
                pl[(e + hi * 8) * 40 + nt * 16 + l16] = (__bf16)s[nt][e];
        asm volatile("s_wait_dscnt 0x0" ::: "memory");
        const __bf16* pp = pl + l16 * 40 + hi * 16;
        v16bf pf = cat8(*(const v8bf*)pp, *(const v8bf*)(pp + 8));
#pragma unroll
        for (int nt = 0; nt < 4; ++nt) {
            const __bf16* vp = Vp + (size_t)(nt * 16 + l16) * TT + jb + hi * 16;
            v16bf vf = *(const v16bf*)vp;
            o[nt] = wmma_bf16(pf, vf, o[nt]);
        }
    }
    const int b = bh >> 4, h = bh & 15;
#pragma unroll
    for (int e = 0; e < 8; ++e) {
        float inv = 1.0f / l[e];
        int t = qbase + e + hi * 8;
#pragma unroll
        for (int nt = 0; nt < 4; ++nt) {
            int d = nt * 16 + l16;
            Y[((size_t)b * TT + t) * CC + h * HS + d] = (__bf16)(o[nt][e] * inv);
        }
    }
}

// ---------------------------------------------------------------------------
// Kernel 3: out = y @ w_proj + b_proj  (A, B both bf16 in ws; fp32 out)
// ---------------------------------------------------------------------------
__global__ __launch_bounds__(256) void k_proj(const __bf16* __restrict__ Y,
                                              const __bf16* __restrict__ Wpt,
                                              const float* __restrict__ bias,
                                              float* __restrict__ out) {
    __shared__ __attribute__((aligned(32))) __bf16 As[128 * LSTR];
    __shared__ __attribute__((aligned(32))) __bf16 Bs[128 * LSTR];
    const int tid  = threadIdx.x;
    const int lane = tid & 31;
    const int wave = tid >> 5;
    const int l16  = lane & 15;
    const int hi   = lane >> 4;
    const int bm = blockIdx.y * 128;
    const int bn = blockIdx.x * 128;
    const int wm = (wave & 1) * 64;
    const int wn = (wave >> 1) * 32;

    const float bias0 = bias[bn + wn + l16];
    const float bias1 = bias[bn + wn + 16 + l16];

    v8f acc[4][2];
#pragma unroll
    for (int i = 0; i < 4; ++i)
#pragma unroll
        for (int j = 0; j < 2; ++j)
#pragma unroll
            for (int e = 0; e < 8; ++e) acc[i][j][e] = 0.0f;

    struct Stage { v16bf a[2]; v16bf b[2]; };
    Stage cur, nxt;

    auto ld = [&](Stage& s, int kb) {
#pragma unroll
        for (int j = 0; j < 2; ++j) {
            int idx = tid + j * 256;
            int row = idx >> 2, c16 = (idx & 3) << 4;
            s.a[j] = *(const v16bf*)(Y   + (size_t)(bm + row) * CC + kb + c16);
            s.b[j] = *(const v16bf*)(Wpt + (size_t)(bn + row) * CC + kb + c16);
        }
    };
    auto st = [&](Stage& s) {
#pragma unroll
        for (int j = 0; j < 2; ++j) {
            int idx = tid + j * 256;
            int row = idx >> 2, c16 = (idx & 3) << 4;
            *(v16bf*)&As[row * LSTR + c16] = s.a[j];
            *(v16bf*)&Bs[row * LSTR + c16] = s.b[j];
        }
    };

    ld(cur, 0);
    for (int it = 0; it < CC / KBLK; ++it) {
        st(cur);
        __syncthreads();
        if (it < CC / KBLK - 1) ld(nxt, (it + 1) * KBLK);
#pragma unroll
        for (int ks = 0; ks < 2; ++ks) {
            v16bf af[4], bfr[2];
#pragma unroll
            for (int mt = 0; mt < 4; ++mt) {
                const __bf16* p = &As[(wm + mt * 16 + l16) * LSTR + ks * 32 + hi * 16];
                af[mt] = cat8(*(const v8bf*)p, *(const v8bf*)(p + 8));
            }
#pragma unroll
            for (int nt = 0; nt < 2; ++nt)
                bfr[nt] = *(const v16bf*)&Bs[(wn + nt * 16 + l16) * LSTR + ks * 32 + hi * 16];
#pragma unroll
            for (int mt = 0; mt < 4; ++mt)
#pragma unroll
                for (int nt = 0; nt < 2; ++nt)
                    acc[mt][nt] = wmma_bf16(af[mt], bfr[nt], acc[mt][nt]);
        }
        __syncthreads();
        cur = nxt;
    }

#pragma unroll
    for (int mt = 0; mt < 4; ++mt)
#pragma unroll
        for (int nt = 0; nt < 2; ++nt)
#pragma unroll
            for (int e = 0; e < 8; ++e) {
                int grow = bm + wm + mt * 16 + e + hi * 8;
                int gcol = bn + wn + nt * 16 + l16;
                out[(size_t)grow * CC + gcol] = acc[mt][nt][e] + (nt ? bias1 : bias0);
            }
}

// ---------------------------------------------------------------------------
extern "C" void kernel_launch(void* const* d_in, const int* in_sizes, int n_in,
                              void* d_out, int out_size, void* d_ws, size_t ws_size,
                              hipStream_t stream) {
    const float* x      = (const float*)d_in[0];
    const float* w_attn = (const float*)d_in[1];
    const float* b_attn = (const float*)d_in[2];
    const float* w_proj = (const float*)d_in[3];
    const float* b_proj = (const float*)d_in[4];
    float* out = (float*)d_out;

    char* ws = (char*)d_ws;
    const size_t MB = 1024 * 1024;
    __bf16* Wat = (__bf16*)(ws);                 // 6 MiB  [3072][1024]
    __bf16* Wpt = (__bf16*)(ws + 6 * MB);        // 2 MiB  [1024][1024]
    __bf16* Qb  = (__bf16*)(ws + 8 * MB);        // 8 MiB  [B,H,T,64]
    __bf16* Kb  = (__bf16*)(ws + 16 * MB);       // 8 MiB  [B,H,T,64]
    __bf16* Vt  = (__bf16*)(ws + 24 * MB);       // 8 MiB  [B,H,64,T]
    __bf16* Yb  = (__bf16*)(ws + 32 * MB);       // 8 MiB  [B*T, C]

    k_convw<<<(N1 * CC + CC * CC) / 256, 256, 0, stream>>>(w_attn, w_proj, Wat, Wpt);

    dim3 g1(N1 / 128, MM / 128);      // 24 x 32
    k_qkv<<<g1, 256, 0, stream>>>(x, Wat, b_attn, Qb, Kb, Vt);

    dim3 g2(TT / 16 / 4, BB * HH);    // 32 x 32, 4 waves/block
    k_attn<<<g2, 128, 0, stream>>>(Qb, Kb, Vt, Yb);

    dim3 g3(CC / 128, MM / 128);      // 8 x 32
    k_proj<<<g3, 256, 0, stream>>>(Yb, w_proj ? Wpt : Wpt, b_proj, out);
}
